// HistoricalCrossAttention_34832184771344
// MI455X (gfx1250) — matrix operand order repaired and checked by reading, not compiled, not run on hardware
//
#include <hip/hip_runtime.h>
#include <stdint.h>

// ---------------------------------------------------------------------------
// HistoricalCrossAttention for MI455X (gfx1250, wave32, WMMA + TDM)
// B=4, T=2048, D=512, H=8, Hd=64
// fp32 inputs -> bf16 operands, all matmuls on v_wmma_f32_16x16x32_bf16
// (fp32 accumulate). Flash attention stages K/V tiles into LDS with the
// Tensor Data Mover (tensor_load_to_lds, TENSORcnt double buffering).
// ---------------------------------------------------------------------------

typedef __attribute__((ext_vector_type(16))) __bf16    v16bf;
typedef __attribute__((ext_vector_type(8)))  float     v8f;
typedef __attribute__((ext_vector_type(4)))  uint32_t  v4u;
typedef __attribute__((ext_vector_type(8)))  int       v8i;
typedef __attribute__((ext_vector_type(4)))  int       v4i;

struct Frag { uint4 lo, hi; };            // 32 bytes = 16 bf16 = one WMMA operand
static_assert(sizeof(Frag) == 32, "frag size");

static __device__ __forceinline__ v16bf as_bf16(const Frag& f) {
    return __builtin_bit_cast(v16bf, f);
}

static __device__ __forceinline__ v8f vzero() {
    v8f v;
    #pragma unroll
    for (int i = 0; i < 8; i++) v[i] = 0.0f;
    return v;
}

static __device__ __forceinline__ uint16_t f2bf(float x) {   // RNE fp32->bf16
    uint32_t u = __float_as_uint(x);
    u += 0x7FFFu + ((u >> 16) & 1u);
    return (uint16_t)(u >> 16);
}

#define WMMA_BF16(a, b, c) \
    __builtin_amdgcn_wmma_f32_16x16x32_bf16(false, (a), false, (b), (short)0, (c), false, false)

// ---------------------------------------------------------------------------
// TDM: DMA a 64x64 bf16 2D tile (row stride stride0 elems) from global -> LDS.
// D# fields packed per CDNA5 ISA 8.3/8.4: group0 = {count, lds_addr,
// global_addr, type=2}; group1 = {data_size=2B, tensor dims = tile dims
// (tile always fully in-bounds), tile 64x64, dim0 stride}.
// ---------------------------------------------------------------------------
static __device__ __forceinline__ void tdm_load_tile64x64(uint64_t gaddr,
                                                          uint32_t lds_addr,
                                                          uint32_t stride0) {
    v4u g0;
    g0[0] = 1u;                                               // count=1, user D#
    g0[1] = lds_addr;                                         // LDS byte address
    g0[2] = (uint32_t)gaddr;                                  // global addr [31:0]
    g0[3] = ((uint32_t)(gaddr >> 32) & 0x01FFFFFFu) | (2u << 30); // [56:32] | type=2
    v8i g1;
    g1[0] = (int)(1u << 16);          // wg_mask=0 (not in cluster), data_size=2B
    g1[1] = (int)(64u << 16);         // atomic_barrier=0 | tensor_dim0[15:0]=64
    g1[2] = (int)(64u << 16);         // tensor_dim0[31:16]=0 | tensor_dim1[15:0]=64
    g1[3] = (int)(64u << 16);         // tensor_dim1[31:16]=0 | tile_dim0=64
    g1[4] = 64;                       // tile_dim1=64 | tile_dim2=0
    g1[5] = (int)stride0;             // tensor_dim0_stride[31:0] (elements)
    g1[6] = 0;                        // stride0[47:32] | stride1[15:0]
    g1[7] = 0;                        // stride1[47:16]
    v4i g2; g2[0] = 0; g2[1] = 0; g2[2] = 0; g2[3] = 0;       // 2D: groups 2/3 unused
    v4i g3 = g2;
#if defined(__clang_major__) && (__clang_major__ >= 23)
    v8i g4;
    #pragma unroll
    for (int i = 0; i < 8; i++) g4[i] = 0;
    __builtin_amdgcn_tensor_load_to_lds(g0, g1, g2, g3, g4, 0);
#else
    __builtin_amdgcn_tensor_load_to_lds(g0, g1, g2, g3, 0);
#endif
}

static __device__ __forceinline__ uint32_t lds_offset(const void* p) {
    // flat addr low 32 bits == LDS byte address (CDNA5 aperture mapping)
    return (uint32_t)(uintptr_t)p;
}

// ---------------------------------------------------------------------------
// Stage 1: fp32 -> bf16 conversion (vectorized, bandwidth-trivial)
// ---------------------------------------------------------------------------
__global__ __launch_bounds__(256)
void cvt4_f32_bf16(const float4* __restrict__ src, uint2* __restrict__ dst, int n4) {
    int i = blockIdx.x * blockDim.x + threadIdx.x;
    if (i < n4) {
        float4 v = src[i];
        uint2 o;
        o.x = (uint32_t)f2bf(v.x) | ((uint32_t)f2bf(v.y) << 16);
        o.y = (uint32_t)f2bf(v.z) | ((uint32_t)f2bf(v.w) << 16);
        dst[i] = o;
    }
}

// ---------------------------------------------------------------------------
// Stage 2/4: out = A(bf16,[8192][512]) * W(bf16,[512][512] rows=out cols)^T + b
// 128x128 block tile, 8 waves x (64M x 32N), K step 32 -> 8 wmma/k-step/wave.
// mode 0: bf16 per-head [B,H,T,64] (Q,K); mode 1: bf16 [B,H,64,T] (V^T);
// mode 2: fp32 [8192][512] (final output)
// ---------------------------------------------------------------------------
__global__ __launch_bounds__(256)
void gemm_xwT(const uint16_t* __restrict__ A, const uint16_t* __restrict__ W,
              const float* __restrict__ bias, void* __restrict__ outp, int mode) {
    const int lane = threadIdx.x & 31;
    const int wave = threadIdx.x >> 5;
    const int wm   = wave & 1;
    const int wn   = wave >> 1;
    const int half = lane >> 4;
    const int l16  = lane & 15;

    const int rowA0 = blockIdx.x * 128 + wm * 64;
    const int colB0 = blockIdx.y * 128 + wn * 32;

    v8f C[4][2];
    #pragma unroll
    for (int i = 0; i < 4; i++)
        #pragma unroll
        for (int j = 0; j < 2; j++) C[i][j] = vzero();

    const char* Ab = (const char*)A;    // row stride 1024 B
    const char* Wb = (const char*)W;    // row stride 1024 B

    for (int k = 0; k < 512; k += 32) {
        Frag af[4];
        #pragma unroll
        for (int mt = 0; mt < 4; mt++) {
            const char* p = Ab + (size_t)(rowA0 + mt * 16 + l16) * 1024 + k * 2 + half * 16;
            af[mt].lo = *(const uint4*)p;
            af[mt].hi = *(const uint4*)(p + 32);
        }
        Frag bfr[2];
        #pragma unroll
        for (int nt = 0; nt < 2; nt++) {
            const char* p = Wb + (size_t)(colB0 + nt * 16 + l16) * 1024 + k * 2 + half * 16;
            bfr[nt].lo = *(const uint4*)p;
            bfr[nt].hi = *(const uint4*)(p + 32);
        }
        #pragma unroll
        for (int mt = 0; mt < 4; mt++)
            #pragma unroll
            for (int nt = 0; nt < 2; nt++)
                C[mt][nt] = WMMA_BF16(as_bf16(af[mt]), as_bf16(bfr[nt]), C[mt][nt]);
    }

    #pragma unroll
    for (int nt = 0; nt < 2; nt++) {
        const int ncol = colB0 + nt * 16 + l16;
        const float bv = bias[ncol];
        const int h  = ncol >> 6;
        const int hd = ncol & 63;
        #pragma unroll
        for (int mt = 0; mt < 4; mt++) {
            #pragma unroll
            for (int r = 0; r < 8; r++) {
                const int mrow = rowA0 + mt * 16 + r + 8 * half;
                const float v = C[mt][nt][r] + bv;
                if (mode == 2) {
                    ((float*)outp)[(size_t)mrow * 512 + ncol] = v;
                } else {
                    const int b = mrow >> 11;
                    const int t = mrow & 2047;
                    uint16_t* o = (uint16_t*)outp;
                    if (mode == 0)
                        o[(((size_t)(b * 8 + h)) * 2048 + t) * 64 + hd] = f2bf(v);
                    else
                        o[(((size_t)(b * 8 + h)) * 64 + hd) * 2048 + t] = f2bf(v);
                }
            }
        }
    }
}

// ---------------------------------------------------------------------------
// Stage 3: flash attention per (b,h). 8 waves x 16 q-rows, 64 keys/step.
// K/V tiles (64x64 bf16, 8 KB each) DMA'd to LDS by the TDM, double-buffered
// on TENSORcnt. Per step/wave: 8 wmma for S = Q K^T, online softmax (shfl
// width-16 row reductions), P relayout through wave-private LDS, 8 wmma for
// O += P V (V held transposed so K-dim spans are contiguous).
// ---------------------------------------------------------------------------
__global__ __launch_bounds__(256)
void flash_attn(const uint16_t* __restrict__ Q, const uint16_t* __restrict__ K,
                const uint16_t* __restrict__ VT, uint16_t* __restrict__ O) {
    __shared__ __align__(16) uint16_t Kt[2][64][64];     // 16 KB, TDM dest
    __shared__ __align__(16) uint16_t Vt[2][64][64];     // 16 KB, TDM dest
    __shared__ __align__(16) uint16_t Plds[8][16][64];   // 16 KB, per-wave P

    const int lane = threadIdx.x & 31;
    const int wave = threadIdx.x >> 5;
    const int half = lane >> 4;
    const int l16  = lane & 15;

    const int bh = blockIdx.y;          // b*8 + h
    const int b  = bh >> 3, h = bh & 7;
    const int q0 = blockIdx.x * 128 + wave * 16;

    const char*    Qb    = (const char*)(Q + (size_t)bh * 2048 * 64);
    const uint64_t Kbase = (uint64_t)(uintptr_t)(K  + (size_t)bh * 2048 * 64); // row 128 B
    const uint64_t Vbase = (uint64_t)(uintptr_t)(VT + (size_t)bh * 64 * 2048); // row 4096 B

    // Q A-frags for K-steps kk=0,32 (row = q0+l16, 64 bf16 per row)
    Frag qf[2];
    {
        const char* p = Qb + (size_t)(q0 + l16) * 128 + half * 16;
        qf[0].lo = *(const uint4*)p;        qf[0].hi = *(const uint4*)(p + 32);
        qf[1].lo = *(const uint4*)(p + 64); qf[1].hi = *(const uint4*)(p + 96);
    }

    float mstate[8], lstate[8];
    #pragma unroll
    for (int r = 0; r < 8; r++) { mstate[r] = -INFINITY; lstate[r] = 0.0f; }
    v8f Oc[4];
    #pragma unroll
    for (int nt = 0; nt < 4; nt++) Oc[nt] = vzero();

    // prologue: DMA tiles for j=0 into buffer 0 (one wave drives the TDM)
    if (wave == 0) {
        tdm_load_tile64x64(Kbase,           lds_offset(&Kt[0][0][0]), 64);
        tdm_load_tile64x64(Vbase,           lds_offset(&Vt[0][0][0]), 2048);
    }

    for (int j = 0; j < 32; j++) {      // 64 keys per iteration
        const int cur = j & 1;
        if (wave == 0) __builtin_amdgcn_s_wait_tensorcnt(0);   // buf[cur] landed
        __syncthreads();   // data visible to all waves; prior buf[cur^1] reads done

        if (wave == 0 && j + 1 < 32) {   // prefetch next tiles into other buffer
            tdm_load_tile64x64(Kbase + (uint64_t)(j + 1) * 64 * 128,
                               lds_offset(&Kt[cur ^ 1][0][0]), 64);
            tdm_load_tile64x64(Vbase + (uint64_t)(j + 1) * 128,
                               lds_offset(&Vt[cur ^ 1][0][0]), 2048);
        }

        // ---- S = Q * K^T : 4 key sub-tiles of 16, K-dim 64 split 32+32 ----
        v8f S[4];
        #pragma unroll
        for (int t = 0; t < 4; t++) S[t] = vzero();
        #pragma unroll
        for (int t = 0; t < 4; t++) {
            const uint16_t* krow = &Kt[cur][t * 16 + l16][0];   // col = key
            #pragma unroll
            for (int kk = 0; kk < 2; kk++) {
                Frag kf;
                kf.lo = *(const uint4*)(krow + kk * 32 + 8 * half);
                kf.hi = *(const uint4*)(krow + kk * 32 + 16 + 8 * half);
                S[t] = WMMA_BF16(as_bf16(qf[kk]), as_bf16(kf), S[t]);
            }
        }

        // ---- online softmax over 64 new keys ----
        float mnew[8], scl[8];
        #pragma unroll
        for (int r = 0; r < 8; r++) {
            float s0 = S[0][r] * 0.125f, s1 = S[1][r] * 0.125f;   // 1/sqrt(64)
            float s2 = S[2][r] * 0.125f, s3 = S[3][r] * 0.125f;
            S[0][r] = s0; S[1][r] = s1; S[2][r] = s2; S[3][r] = s3;
            float mx = fmaxf(fmaxf(s0, s1), fmaxf(s2, s3));
            #pragma unroll
            for (int off = 1; off < 16; off <<= 1)
                mx = fmaxf(mx, __shfl_xor(mx, off, 16));
            mnew[r] = fmaxf(mstate[r], mx);
            scl[r]  = __expf(mstate[r] - mnew[r]);
            mstate[r] = mnew[r];
        }
        #pragma unroll
        for (int r = 0; r < 8; r++) {
            float p0 = __expf(S[0][r] - mnew[r]);
            float p1 = __expf(S[1][r] - mnew[r]);
            float p2 = __expf(S[2][r] - mnew[r]);
            float p3 = __expf(S[3][r] - mnew[r]);
            float sm = (p0 + p1) + (p2 + p3);
            #pragma unroll
            for (int off = 1; off < 16; off <<= 1)
                sm += __shfl_xor(sm, off, 16);
            lstate[r] = lstate[r] * scl[r] + sm;
            // C-layout -> row-major bf16 P in wave-private LDS (in-order DS)
            uint16_t* prow = &Plds[wave][r + 8 * half][0];
            prow[ 0 + l16] = f2bf(p0);
            prow[16 + l16] = f2bf(p1);
            prow[32 + l16] = f2bf(p2);
            prow[48 + l16] = f2bf(p3);
        }
        #pragma unroll
        for (int nt = 0; nt < 4; nt++)
            #pragma unroll
            for (int r = 0; r < 8; r++)
                Oc[nt][r] *= scl[r];

        // ---- O += P(16x64) * V(64x64), keys split 32+32 ----
        #pragma unroll
        for (int kk2 = 0; kk2 < 2; kk2++) {
            Frag pf;
            const uint16_t* prow = &Plds[wave][l16][0];
            pf.lo = *(const uint4*)(prow + kk2 * 32 + 8 * half);
            pf.hi = *(const uint4*)(prow + kk2 * 32 + 16 + 8 * half);
            #pragma unroll
            for (int nt = 0; nt < 4; nt++) {
                const uint16_t* vrow = &Vt[cur][nt * 16 + l16][0];  // row = hd
                Frag vf;
                vf.lo = *(const uint4*)(vrow + kk2 * 32 + 8 * half);
                vf.hi = *(const uint4*)(vrow + kk2 * 32 + 16 + 8 * half);
                Oc[nt] = WMMA_BF16(as_bf16(pf), as_bf16(vf), Oc[nt]);
            }
        }
    }

    // normalize and store to O[B][T][512] (bf16, feeds final projection)
    float inv[8];
    #pragma unroll
    for (int r = 0; r < 8; r++) inv[r] = 1.0f / lstate[r];
    #pragma unroll
    for (int nt = 0; nt < 4; nt++) {
        const int col = h * 64 + nt * 16 + l16;
        #pragma unroll
        for (int r = 0; r < 8; r++) {
            const int row = q0 + r + 8 * half;
            O[((size_t)b * 2048 + row) * 512 + col] = f2bf(Oc[nt][r] * inv[r]);
        }
    }
}

// ---------------------------------------------------------------------------
// Host-side orchestration
// ---------------------------------------------------------------------------
extern "C" void kernel_launch(void* const* d_in, const int* in_sizes, int n_in,
                              void* d_out, int out_size, void* d_ws, size_t ws_size,
                              hipStream_t stream) {
    (void)in_sizes; (void)n_in; (void)out_size; (void)ws_size;

    const float* xc = (const float*)d_in[0];
    const float* xp = (const float*)d_in[1];
    const float* wq = (const float*)d_in[2];
    const float* bq = (const float*)d_in[3];
    const float* wk = (const float*)d_in[4];
    const float* bk = (const float*)d_in[5];
    const float* wv = (const float*)d_in[6];
    const float* bv = (const float*)d_in[7];
    const float* wo = (const float*)d_in[8];
    const float* bo = (const float*)d_in[9];

    const size_t NX = (size_t)8192 * 512;   // x / Q / K / V / O element count
    const size_t NW = (size_t)512 * 512;    // weight element count

    uint16_t* xcb = (uint16_t*)d_ws;
    uint16_t* xpb = xcb + NX;
    uint16_t* wqb = xpb + NX;
    uint16_t* wkb = wqb + NW;
    uint16_t* wvb = wkb + NW;
    uint16_t* wob = wvb + NW;
    uint16_t* Qb  = wob + NW;   // [4][8][2048][64]
    uint16_t* Kb  = Qb  + NX;   // [4][8][2048][64]
    uint16_t* VTb = Kb  + NX;   // [4][8][64][2048]
    uint16_t* Ob  = VTb + NX;   // [4][2048][512]

    {
        const int nx4 = (int)(NX / 4), nw4 = (int)(NW / 4);
        dim3 blk(256);
        cvt4_f32_bf16<<<dim3((nx4 + 255) / 256), blk, 0, stream>>>((const float4*)xc, (uint2*)xcb, nx4);
        cvt4_f32_bf16<<<dim3((nx4 + 255) / 256), blk, 0, stream>>>((const float4*)xp, (uint2*)xpb, nx4);
        cvt4_f32_bf16<<<dim3((nw4 + 255) / 256), blk, 0, stream>>>((const float4*)wq, (uint2*)wqb, nw4);
        cvt4_f32_bf16<<<dim3((nw4 + 255) / 256), blk, 0, stream>>>((const float4*)wk, (uint2*)wkb, nw4);
        cvt4_f32_bf16<<<dim3((nw4 + 255) / 256), blk, 0, stream>>>((const float4*)wv, (uint2*)wvb, nw4);
        cvt4_f32_bf16<<<dim3((nw4 + 255) / 256), blk, 0, stream>>>((const float4*)wo, (uint2*)wob, nw4);
    }

    dim3 ggrid(64, 4), gblk(256);
    gemm_xwT<<<ggrid, gblk, 0, stream>>>(xcb, wqb, bq, Qb,  0);
    gemm_xwT<<<ggrid, gblk, 0, stream>>>(xpb, wkb, bk, Kb,  0);
    gemm_xwT<<<ggrid, gblk, 0, stream>>>(xpb, wvb, bv, VTb, 1);

    flash_attn<<<dim3(16, 32), dim3(256), 0, stream>>>(Qb, Kb, VTb, Ob);

    gemm_xwT<<<ggrid, gblk, 0, stream>>>(Ob, wob, bo, d_out, 2);
}
